// SPINN_7834020348669
// MI455X (gfx1250) — matrix operand order repaired and checked by reading, not compiled, not run on hardware
//
#include <hip/hip_runtime.h>
#include <hip/hip_bf16.h>
#include <stdint.h>

// Problem constants (from reference): B=64, L=48, D=256, TK=128, T=2L-1=95
constexpr int kB  = 64;
constexpr int kL  = 48;
constexpr int kD  = 256;
constexpr int kTK = 128;
constexpr int kT  = 2 * kL - 1;   // 95
constexpr int kG4 = 4 * kTK;      // 512
constexpr int kG5 = 5 * kD;       // 1280

constexpr int kThreads = 512;     // 16 wave32 waves on one WGP

typedef __attribute__((ext_vector_type(16))) __bf16 bf16x16;
typedef __attribute__((ext_vector_type(8)))  __bf16 bf16x8;
typedef __attribute__((ext_vector_type(8)))  float  f32x8;

__device__ __forceinline__ float sigm(float x) { return 1.0f / (1.0f + __expf(-x)); }
__device__ __forceinline__ int imax0(int x) { return x < 0 ? 0 : x; }

// Workgroup-scope release+acquire: all waves run on one WGP, no need for
// device-scope global_wb/global_inv at phase boundaries.
__device__ __forceinline__ void wg_fence_barrier() {
  __builtin_amdgcn_fence(__ATOMIC_SEQ_CST, "workgroup");
  __syncthreads();
}

__device__ __forceinline__ f32x8 wmma_bf16(bf16x16 a, bf16x16 b, f32x8 c) {
  // D = A(16x32 bf16) * B(32x16 bf16) + C(16x16 f32)
  return __builtin_amdgcn_wmma_f32_16x16x32_bf16(false, a, false, b, (short)0, c,
                                                 false, false);
}

// A-operand tile (16x32 bf16) from a row-major [rows x ldk] bf16 buffer.
// Layout: lanes 0-15 hold row (lane), K chunks {0..7, 16..23};
// lanes 16-31 hold row (lane-16), K chunks {8..15, 24..31}.
__device__ __forceinline__ bf16x16 load_a(const __bf16* __restrict__ base, int ldk,
                                          int m0, int k0, int lane) {
  int row  = m0 + (lane & 15);
  int half = lane >> 4;
  const __bf16* p = base + row * ldk + k0 + half * 8;
  bf16x8 lo = *reinterpret_cast<const bf16x8*>(p);
  bf16x8 hi = *reinterpret_cast<const bf16x8*>(p + 16);
  return __builtin_shufflevector(lo, hi, 0,1,2,3,4,5,6,7,8,9,10,11,12,13,14,15);
}

// B-operand tile (32x16 bf16) from pre-packed operand blocks:
// block = 32 lanes x 16 bf16 contiguous per lane (1024 B).
__device__ __forceinline__ bf16x16 load_b(const __bf16* __restrict__ pack,
                                          int blockId, int lane) {
  const __bf16* p = pack + blockId * 512 + lane * 16;
  bf16x8 lo = *reinterpret_cast<const bf16x8*>(p);
  bf16x8 hi = *reinterpret_cast<const bf16x8*>(p + 8);
  return __builtin_shufflevector(lo, hi, 0,1,2,3,4,5,6,7,8,9,10,11,12,13,14,15);
}

// Pack a row-major f32 weight (K x N) into bf16 WMMA B-operand blocks.
// blockId = nTile * (K/32) + kTile; within block: lane c=(l&15) holds column
// nTile*16+c, K = kTile*32 + (l>>4)*8 + (j&7) + (j>>3)*16.
__global__ void pack_weights(const float* __restrict__ W, __bf16* __restrict__ dst,
                             int K, int N) {
  int idx = blockIdx.x * blockDim.x + threadIdx.x;
  if (idx >= K * N) return;
  int blockId = idx >> 9;          // / 512
  int rem     = idx & 511;
  int laneI   = rem >> 4;
  int j       = rem & 15;
  int kTiles  = K >> 5;
  int nT      = blockId / kTiles;
  int kT2     = blockId % kTiles;
  int n = nT * 16 + (laneI & 15);
  int k = kT2 * 32 + (laneI >> 4) * 8 + (j & 7) + (j >> 3) * 16;
  dst[idx] = (__bf16)W[k * N + n];
}

__global__ void zero_f32(float* __restrict__ p, int n) {
  int i = blockIdx.x * blockDim.x + threadIdx.x;
  if (i < n) p[i] = 0.0f;
}

__global__ __launch_bounds__(kThreads, 1)
void spinn_persistent(const float* __restrict__ buf_h, const float* __restrict__ buf_c,
                      const int* __restrict__ trans,
                      const float* __restrict__ bl, const float* __restrict__ bLv,
                      const __bf16* __restrict__ pWb, const __bf16* __restrict__ pW1,
                      const __bf16* __restrict__ pW2, const __bf16* __restrict__ pWl,
                      const __bf16* __restrict__ pWL, const __bf16* __restrict__ pWR,
                      const __bf16* __restrict__ pWT,
                      float* __restrict__ stackF,
                      __bf16* __restrict__ tbh, __bf16* __restrict__ s1h,
                      __bf16* __restrict__ s2h, __bf16* __restrict__ th2p,
                      float* __restrict__ out) {
  __shared__ float tc_s[kB * kTK];   // tracker cell state, 32 KB
  __shared__ int   sp_s[kB];
  __shared__ int   bp_s[kB];

  const int tid  = threadIdx.x;
  const int wave = tid >> 5;
  const int lane = tid & 31;

  for (int i = tid; i < kB * kTK; i += kThreads) tc_s[i] = 0.0f;
  if (tid < kB) { sp_s[tid] = 0; bp_s[tid] = 0; }
  __syncthreads();

  for (int t = 0; t < kT; ++t) {
    const __bf16* thCur = th2p + ((t & 1) * kB * kTK);
    __bf16*       thNxt = th2p + (((t + 1) & 1) * kB * kTK);

    // ---------------- Phase G: gather A operands (bf16 staging) -------------
    for (int idx = tid; idx < kB * kD; idx += kThreads) {
      int b = idx >> 8;            // / kD
      int d = idx & (kD - 1);
      int spb = sp_s[b];
      int bpb = bp_s[b];
      float tb = (bpb < kL) ? buf_h[(b * kL + bpb) * kD + d] : 0.0f;
      float v1 = (spb >= 1) ? stackF[((b * kL) + (spb - 1)) * (2 * kD) + d] : 0.0f;
      float v2 = (spb >= 2) ? stackF[((b * kL) + (spb - 2)) * (2 * kD) + d] : 0.0f;
      tbh[idx] = (__bf16)tb;
      s1h[idx] = (__bf16)v1;
      s2h[idx] = (__bf16)v2;
    }
    wg_fence_barrier();

    // ------------- Phase A: tracker gates + LSTM (32 jobs, 2/wave) ---------
    for (int jj = 0; jj < 2; ++jj) {
      int job = wave * 2 + jj;            // 0..31
      int mt = job >> 3;                  // batch tile 0..3
      int nc = job & 7;                   // 16-col chunk of TK
      int m0 = mt * 16;
      f32x8 g0 = {}, g1 = {}, g2 = {}, g3 = {};
#pragma unroll 2
      for (int kt = 0; kt < 8; ++kt) {
        bf16x16 av = load_a(tbh, kD, m0, kt * 32, lane);
        g0 = wmma_bf16(av, load_b(pWb, (0 * 8 + nc) * 8 + kt, lane), g0);
        g1 = wmma_bf16(av, load_b(pWb, (1 * 8 + nc) * 8 + kt, lane), g1);
        g2 = wmma_bf16(av, load_b(pWb, (2 * 8 + nc) * 8 + kt, lane), g2);
        g3 = wmma_bf16(av, load_b(pWb, (3 * 8 + nc) * 8 + kt, lane), g3);
      }
#pragma unroll 2
      for (int kt = 0; kt < 8; ++kt) {
        bf16x16 av = load_a(s1h, kD, m0, kt * 32, lane);
        g0 = wmma_bf16(av, load_b(pW1, (0 * 8 + nc) * 8 + kt, lane), g0);
        g1 = wmma_bf16(av, load_b(pW1, (1 * 8 + nc) * 8 + kt, lane), g1);
        g2 = wmma_bf16(av, load_b(pW1, (2 * 8 + nc) * 8 + kt, lane), g2);
        g3 = wmma_bf16(av, load_b(pW1, (3 * 8 + nc) * 8 + kt, lane), g3);
      }
#pragma unroll 2
      for (int kt = 0; kt < 8; ++kt) {
        bf16x16 av = load_a(s2h, kD, m0, kt * 32, lane);
        g0 = wmma_bf16(av, load_b(pW2, (0 * 8 + nc) * 8 + kt, lane), g0);
        g1 = wmma_bf16(av, load_b(pW2, (1 * 8 + nc) * 8 + kt, lane), g1);
        g2 = wmma_bf16(av, load_b(pW2, (2 * 8 + nc) * 8 + kt, lane), g2);
        g3 = wmma_bf16(av, load_b(pW2, (3 * 8 + nc) * 8 + kt, lane), g3);
      }
      if (t > 0) {
#pragma unroll 2
        for (int kt = 0; kt < 4; ++kt) {
          bf16x16 av = load_a(thCur, kTK, m0, kt * 32, lane);
          g0 = wmma_bf16(av, load_b(pWl, (0 * 8 + nc) * 4 + kt, lane), g0);
          g1 = wmma_bf16(av, load_b(pWl, (1 * 8 + nc) * 4 + kt, lane), g1);
          g2 = wmma_bf16(av, load_b(pWl, (2 * 8 + nc) * 4 + kt, lane), g2);
          g3 = wmma_bf16(av, load_b(pWl, (3 * 8 + nc) * 4 + kt, lane), g3);
        }
      }
      // C layout: VGPR v, lanes 0-15 -> M=v, lanes 16-31 -> M=v+8; N=lane&15
      int col   = nc * 16 + (lane & 15);
      int rbase = m0 + ((lane >> 4) << 3);
      float ba = 0.0f, bi = 0.0f, bf = 0.0f, bo = 0.0f;
      if (t > 0) {
        ba = bl[0 * kTK + col]; bi = bl[1 * kTK + col];
        bf = bl[2 * kTK + col]; bo = bl[3 * kTK + col];
      }
#pragma unroll
      for (int v = 0; v < 8; ++v) {
        int b = rbase + v;
        float ga = g0[v] + ba, gi = g1[v] + bi, gf = g2[v] + bf, go = g3[v] + bo;
        float c  = tc_s[b * kTK + col];
        float c2 = tanhf(ga) * sigm(gi) + sigm(gf) * c;
        float h2 = sigm(go) * tanhf(c2);
        tc_s[b * kTK + col] = c2;
        thNxt[b * kTK + col] = (__bf16)h2;
      }
    }
    wg_fence_barrier();

    // --------- Phase B: reduce gates + TreeLSTM + stack write (64 jobs) -----
    for (int jj = 0; jj < 4; ++jj) {
      int job = wave * 4 + jj;            // 0..63
      int mt = job >> 4;                  // 0..3
      int nc = job & 15;                  // 16-col chunk of D
      int m0 = mt * 16;
      f32x8 a0 = {}, a1 = {}, a2 = {}, a3 = {}, a4 = {};
#pragma unroll 2
      for (int kt = 0; kt < 8; ++kt) {
        bf16x16 av = load_a(s2h, kD, m0, kt * 32, lane);
        a0 = wmma_bf16(av, load_b(pWL, (0 * 16 + nc) * 8 + kt, lane), a0);
        a1 = wmma_bf16(av, load_b(pWL, (1 * 16 + nc) * 8 + kt, lane), a1);
        a2 = wmma_bf16(av, load_b(pWL, (2 * 16 + nc) * 8 + kt, lane), a2);
        a3 = wmma_bf16(av, load_b(pWL, (3 * 16 + nc) * 8 + kt, lane), a3);
        a4 = wmma_bf16(av, load_b(pWL, (4 * 16 + nc) * 8 + kt, lane), a4);
      }
#pragma unroll 2
      for (int kt = 0; kt < 8; ++kt) {
        bf16x16 av = load_a(s1h, kD, m0, kt * 32, lane);
        a0 = wmma_bf16(av, load_b(pWR, (0 * 16 + nc) * 8 + kt, lane), a0);
        a1 = wmma_bf16(av, load_b(pWR, (1 * 16 + nc) * 8 + kt, lane), a1);
        a2 = wmma_bf16(av, load_b(pWR, (2 * 16 + nc) * 8 + kt, lane), a2);
        a3 = wmma_bf16(av, load_b(pWR, (3 * 16 + nc) * 8 + kt, lane), a3);
        a4 = wmma_bf16(av, load_b(pWR, (4 * 16 + nc) * 8 + kt, lane), a4);
      }
#pragma unroll 2
      for (int kt = 0; kt < 4; ++kt) {
        bf16x16 av = load_a(thNxt, kTK, m0, kt * 32, lane);
        a0 = wmma_bf16(av, load_b(pWT, (0 * 16 + nc) * 4 + kt, lane), a0);
        a1 = wmma_bf16(av, load_b(pWT, (1 * 16 + nc) * 4 + kt, lane), a1);
        a2 = wmma_bf16(av, load_b(pWT, (2 * 16 + nc) * 4 + kt, lane), a2);
        a3 = wmma_bf16(av, load_b(pWT, (3 * 16 + nc) * 4 + kt, lane), a3);
        a4 = wmma_bf16(av, load_b(pWT, (4 * 16 + nc) * 4 + kt, lane), a4);
      }
      int col   = nc * 16 + (lane & 15);
      int rbase = m0 + ((lane >> 4) << 3);
      float ba  = bLv[0 * kD + col];
      float bi  = bLv[1 * kD + col];
      float bf1 = bLv[2 * kD + col];
      float bf2 = bLv[3 * kD + col];
      float bo  = bLv[4 * kD + col];
#pragma unroll
      for (int v = 0; v < 8; ++v) {
        int b   = rbase + v;
        int spb = sp_s[b];
        int bpb = bp_s[b];
        bool shift = (trans[b * kT + t] == 0);
        float ga  = a0[v] + ba;
        float gi  = a1[v] + bi;
        float gf1 = a2[v] + bf1;
        float gf2 = a3[v] + bf2;
        float go  = a4[v] + bo;
        float c1  = (spb >= 2) ? stackF[((b * kL) + (spb - 2)) * (2 * kD) + kD + col] : 0.0f;
        float c2v = (spb >= 1) ? stackF[((b * kL) + (spb - 1)) * (2 * kD) + kD + col] : 0.0f;
        float rc = tanhf(ga) * sigm(gi) + sigm(gf1) * c1 + sigm(gf2) * c2v;
        float rh = sigm(go) * tanhf(rc);
        int pos = shift ? spb : imax0(spb - 2);
        float oh, oc;
        if (shift) {
          if (bpb < kL) {
            oh = buf_h[(b * kL + bpb) * kD + col];
            oc = buf_c[(b * kL + bpb) * kD + col];
          } else { oh = 0.0f; oc = 0.0f; }
        } else { oh = rh; oc = rc; }
        float* dst = stackF + ((b * kL) + pos) * (2 * kD);
        dst[col]      = oh;
        dst[kD + col] = oc;
      }
    }
    wg_fence_barrier();

    // ---------------- Phase U: stack/buffer pointer update ------------------
    if (tid < kB) {
      bool shift = (trans[tid * kT + t] == 0);
      int spb = sp_s[tid];
      sp_s[tid] = shift ? (spb + 1) : imax0(spb - 1);
      bp_s[tid] += shift ? 1 : 0;
    }
    __syncthreads();
  }

  // ---------------------- Final: out = stack[b, sp-1] -----------------------
  for (int idx = tid; idx < kB * 2 * kD; idx += kThreads) {
    int b = idx / (2 * kD);
    int j = idx % (2 * kD);
    int spb  = sp_s[b];
    int slot = imax0(spb - 1);
    if (slot > kL - 1) slot = kL - 1;
    out[idx] = stackF[((b * kL) + slot) * (2 * kD) + j];
  }
}

extern "C" void kernel_launch(void* const* d_in, const int* in_sizes, int n_in,
                              void* d_out, int out_size, void* d_ws, size_t ws_size,
                              hipStream_t stream) {
  (void)in_sizes; (void)n_in; (void)out_size; (void)ws_size;
  const float* buf_h = (const float*)d_in[0];
  const float* buf_c = (const float*)d_in[1];
  const float* Wb    = (const float*)d_in[2];
  const float* W1    = (const float*)d_in[3];
  const float* W2    = (const float*)d_in[4];
  const float* Wl    = (const float*)d_in[5];
  const float* bl    = (const float*)d_in[6];
  const float* WL    = (const float*)d_in[7];
  const float* bL    = (const float*)d_in[8];
  const float* WR    = (const float*)d_in[9];
  const float* WT    = (const float*)d_in[10];
  const int*   tr    = (const int*)d_in[11];
  float* out = (float*)d_out;

  char* base = (char*)d_ws;
  size_t off = 0;
  auto take = [&](size_t bytes) -> char* {
    off = (off + 255) & ~size_t(255);
    char* p = base + off;
    off += bytes;
    return p;
  };
  __bf16* pWb = (__bf16*)take((size_t)kD  * kG4 * 2);
  __bf16* pW1 = (__bf16*)take((size_t)kD  * kG4 * 2);
  __bf16* pW2 = (__bf16*)take((size_t)kD  * kG4 * 2);
  __bf16* pWl = (__bf16*)take((size_t)kTK * kG4 * 2);
  __bf16* pWL = (__bf16*)take((size_t)kD  * kG5 * 2);
  __bf16* pWR = (__bf16*)take((size_t)kD  * kG5 * 2);
  __bf16* pWT = (__bf16*)take((size_t)kTK * kG5 * 2);
  float*  stackF = (float*)take((size_t)kB * kL * 2 * kD * 4);
  __bf16* tbh  = (__bf16*)take((size_t)kB * kD * 2);
  __bf16* s1h  = (__bf16*)take((size_t)kB * kD * 2);
  __bf16* s2h  = (__bf16*)take((size_t)kB * kD * 2);
  __bf16* th2p = (__bf16*)take((size_t)2 * kB * kTK * 2);

  auto packLaunch = [&](const float* W, __bf16* dst, int K, int N) {
    int n = K * N;
    pack_weights<<<(n + 255) / 256, 256, 0, stream>>>(W, dst, K, N);
  };
  packLaunch(Wb, pWb, kD,  kG4);
  packLaunch(W1, pW1, kD,  kG4);
  packLaunch(W2, pW2, kD,  kG4);
  packLaunch(Wl, pWl, kTK, kG4);
  packLaunch(WL, pWL, kD,  kG5);
  packLaunch(WR, pWR, kD,  kG5);
  packLaunch(WT, pWT, kTK, kG5);

  int stackN = kB * kL * 2 * kD;
  zero_f32<<<(stackN + 255) / 256, 256, 0, stream>>>(stackF, stackN);

  spinn_persistent<<<1, kThreads, 0, stream>>>(buf_h, buf_c, tr, bl, bL,
                                               pWb, pW1, pW2, pWl, pWL, pWR, pWT,
                                               stackF, tbh, s1h, s2h, th2p, out);
}
